// FullModelSnowflake_84413287236182
// MI455X (gfx1250) — matrix operand order repaired and checked by reading, not compiled, not run on hardware
//
#include <hip/hip_runtime.h>
#include <hip/hip_bf16.h>
#include <cstdint>
#include <cstddef>

// ---------------------------------------------------------------------------
// Problem constants (match reference)
// ---------------------------------------------------------------------------
#define NB    2
#define NTOK  2048
#define DM    128
#define NH    8
#define DKH   16
#define KNN_K 16
#define NLAY  4
#define NCOARSE 64
#define NSTAGE 7

typedef __attribute__((ext_vector_type(16))) _Float16 v16h;
typedef __attribute__((ext_vector_type(8)))  _Float16 v8h;
typedef __attribute__((ext_vector_type(8)))  float    v8f;

static __device__ __forceinline__ v8f wmma_f16(v16h a, v16h b, v8f c) {
  // D = A(16x32 f16) * B(32x16 f16) + C(16x16 f32)
  return __builtin_amdgcn_wmma_f32_16x16x32_f16(false, a, false, b, (short)0, c,
                                                false, false);
}

static __device__ __forceinline__ v16h cat16(v8h lo, v8h hi) {
  return __builtin_shufflevector(lo, hi, 0, 1, 2, 3, 4, 5, 6, 7,
                                         8, 9, 10, 11, 12, 13, 14, 15);
}

// ---------------------------------------------------------------------------
// Generic WMMA GEMM:  C[b] = act( A[b](MxK) * B[b](KxN) + biases )  (+= opt.)
// Arbitrary element strides for A/B/C (row,col) + batch strides.
// GUARDED=false requires M%64==0, N%64==0, K%32==0 (all hot GEMMs here):
// straight-line staging, no per-element bounds compares/exec-masking.
// 64x64 C-tile per 256-thread workgroup; fp32->f16 staged in LDS (A row-major,
// B transposed so per-lane fragment runs are 16B-contiguous -> ds_load_b128).
// ---------------------------------------------------------------------------
template <bool GUARDED>
__launch_bounds__(256)
__global__ void wmma_gemm_kernel(
    const float* __restrict__ A, int asr, int asc, long long abat,
    const float* __restrict__ Bm, int bsr, int bsc, long long bbat,
    float* __restrict__ C, int csr, int csc, long long cbat, int coff,
    const float* __restrict__ biasCol,
    const float* __restrict__ biasRow,
    const float* __restrict__ rowAdd, long long rabat,
    int M, int Nn, int K, int act, int accum)
{
  __shared__ alignas(16) _Float16 Ah[64][40];   // [row][k]   (80B row stride)
  __shared__ alignas(16) _Float16 Bt[64][40];   // [col][k]   (transposed!)

  const int tid  = threadIdx.x;
  const int wave = tid >> 5;
  const int lane = tid & 31;
  const int bz   = blockIdx.z;

  A  += (long long)bz * abat;
  Bm += (long long)bz * bbat;
  C  += (long long)bz * cbat + coff;
  const float* rAdd = rowAdd ? rowAdd + (long long)bz * rabat : nullptr;

  const int m0 = blockIdx.y * 64;
  const int n0 = blockIdx.x * 64;

  v8f accv[2];
  #pragma unroll
  for (int t = 0; t < 2; ++t)
    #pragma unroll
    for (int r = 0; r < 8; ++r) accv[t][r] = 0.0f;

  // staging coordinates (8 elements per thread, one v8h store each)
  const int sar = tid >> 2;              // A: row 0..63
  const int sac = (tid & 3) << 3;        // A: k base 0,8,16,24
  const int sbc = tid >> 2;              // B: col 0..63
  const int sbr = (tid & 3) << 3;        // B: k base 0,8,16,24

  // per-thread row base pointers (hoisted out of the k loop)
  const float* apRow = A  + (long long)(GUARDED ? 0 : (m0 + sar)) * asr;
  const float* bpCol = Bm + (long long)(GUARDED ? 0 : (n0 + sbc)) * bsc;
  const bool aVec = (!GUARDED) && (asc == 1) && ((asr & 3) == 0);

  for (int k0 = 0; k0 < K; k0 += 32) {
    if (!GUARDED) {
      // ---- A tile: contiguous fast path or strided scalar path ----
      v8h hv;
      if (aVec) {
        const float4 f0 = *(const float4*)(apRow + k0 + sac);
        const float4 f1 = *(const float4*)(apRow + k0 + sac + 4);
        hv[0] = (_Float16)f0.x; hv[1] = (_Float16)f0.y;
        hv[2] = (_Float16)f0.z; hv[3] = (_Float16)f0.w;
        hv[4] = (_Float16)f1.x; hv[5] = (_Float16)f1.y;
        hv[6] = (_Float16)f1.z; hv[7] = (_Float16)f1.w;
      } else {
        #pragma unroll
        for (int j = 0; j < 8; ++j)
          hv[j] = (_Float16)apRow[(long long)(k0 + sac + j) * asc];
      }
      *(v8h*)&Ah[sar][sac] = hv;
      // ---- B tile (strided along k, coalesced along n across lanes) ----
      v8h hb;
      #pragma unroll
      for (int j = 0; j < 8; ++j)
        hb[j] = (_Float16)bpCol[(long long)(k0 + sbr + j) * bsr];
      *(v8h*)&Bt[sbc][sbr] = hb;
    } else {
      // ---- guarded path (ragged dims; only the tiny K=6 GEMM uses this) ----
      {
        const int gm = m0 + sar;
        const int gmc = gm < M ? gm : (M - 1);
        v8h hv;
        #pragma unroll
        for (int j = 0; j < 8; ++j) {
          const int gk = k0 + sac + j;
          const int gkc = gk < K ? gk : (K - 1);
          float vv = A[(long long)gmc * asr + (long long)gkc * asc];
          if (gm >= M || gk >= K) vv = 0.0f;
          hv[j] = (_Float16)vv;
        }
        *(v8h*)&Ah[sar][sac] = hv;
      }
      {
        const int gn = n0 + sbc;
        const int gnc = gn < Nn ? gn : (Nn - 1);
        v8h hv;
        #pragma unroll
        for (int j = 0; j < 8; ++j) {
          const int gk = k0 + sbr + j;
          const int gkc = gk < K ? gk : (K - 1);
          float vv = Bm[(long long)gkc * bsr + (long long)gnc * bsc];
          if (gk >= K || gn >= Nn) vv = 0.0f;
          hv[j] = (_Float16)vv;
        }
        *(v8h*)&Bt[sbc][sbr] = hv;
      }
    }
    __syncthreads();

    // speculative prefetch of next k-tiles (global_prefetch_b8)
    if (!GUARDED && k0 + 32 < K) {
      __builtin_prefetch(apRow + (long long)(k0 + 32 + sac) * asc, 0, 1);
      __builtin_prefetch(bpCol + (long long)(k0 + 32 + sbr) * bsr, 0, 1);
    }

    // ---- 2 WMMA tiles per wave (16 tiles = 4x4 grid of 16x16) ----
    const int k0h = (lane < 16) ? 0 : 8;   // A frag: K=0..7/16..23 vs 8..15/24..31
    const int kb  = (lane < 16) ? 0 : 16;  // unused; B frag uses k0h (same K split as A)
    #pragma unroll
    for (int t = 0; t < 2; ++t) {
      const int tt = wave * 2 + t;
      const int tr = (tt >> 2) << 4;
      const int tc = (tt & 3) << 4;
      const int arow = tr + (lane & 15);
      const int bcol = tc + (lane & 15);
      const v16h af = cat16(*(const v8h*)&Ah[arow][k0h],
                            *(const v8h*)&Ah[arow][16 + k0h]);
      const v16h bf = cat16(*(const v8h*)&Bt[bcol][k0h],
                            *(const v8h*)&Bt[bcol][16 + k0h]);
      accv[t] = wmma_f16(af, bf, accv[t]);
    }
    __syncthreads();
  }

  // ---- epilogue: C layout lane<16 rows 0..7, lane>=16 rows 8..15, col=lane&15
  #pragma unroll
  for (int t = 0; t < 2; ++t) {
    const int tt = wave * 2 + t;
    const int tr = (tt >> 2) << 4;
    const int tc = (tt & 3) << 4;
    const int col = n0 + tc + (lane & 15);
    if (GUARDED && col >= Nn) continue;
    const float bc = biasCol ? biasCol[col] : 0.0f;
    const int rbase = m0 + tr + ((lane < 16) ? 0 : 8);
    #pragma unroll
    for (int r = 0; r < 8; ++r) {
      const int row = rbase + r;
      if (GUARDED && row >= M) continue;
      float vv = accv[t][r] + bc;
      if (biasRow) vv += biasRow[row];
      if (rAdd)    vv += rAdd[row];
      const long long ci = (long long)row * csr + (long long)col * csc;
      if (accum) vv += C[ci];
      if (act == 1) vv = fmaxf(vv, 0.0f);
      C[ci] = vv;
    }
  }
}

// ---------------------------------------------------------------------------
// Flash attention with geometric bias (transposed formulation).
//   S^T = K Q^T  -> each lane owns ONE query column (8 key rows in VGPRs),
//   so softmax max/sum are in-register reductions + ONE shfl_xor(16) with the
//   partner lane holding the other 8 keys.
//   O^T = V^T P^T -> lane owns one query, 8 dk rows; rescale is scalar*v8f and
//   the final store is two global_store_b128 of contiguous dk.
// grid (NTOK/128, NH, NB), 256 threads (8 waves x 16 queries).
// WMMA 16x16x32 with K padded to 32; pad halves live zeroed in LDS so all
// fragment fetches are unconditional ds_load_b128 pairs.
// ---------------------------------------------------------------------------
__launch_bounds__(256)
__global__ void attn_kernel(const float* __restrict__ q,
                            const float* __restrict__ k,
                            const float* __restrict__ v,
                            const float* __restrict__ coords6, // stride 6, xyz
                            const float* __restrict__ alphaP,
                            float* __restrict__ aout)
{
  const int b    = blockIdx.z;
  const int h    = blockIdx.y;
  const int qblk = blockIdx.x * 128;
  const int tid  = threadIdx.x;
  const int wave = tid >> 5;
  const int lane = tid & 31;
  const float alpha = *alphaP;

  __shared__ alignas(16) _Float16 Qt[128][40];    // [query][dk(0..31, hi zero)]
  __shared__ alignas(16) _Float16 Kl[16][40];     // [key][dk(0..31, hi zero)]
  __shared__ alignas(16) _Float16 Vtr[16][40];    // [dk][key(0..31, hi zero)]
  __shared__ alignas(16) _Float16 Pl[8][16][40];  // [wave][query][key(hi zero)]
  __shared__ alignas(16) float4 kc4[16];          // key coords (xyz,0)

  const v8h z8 = {(_Float16)0, (_Float16)0, (_Float16)0, (_Float16)0,
                  (_Float16)0, (_Float16)0, (_Float16)0, (_Float16)0};

  // ---- stage Q tile (128 queries x 16 dk) -> Qt, zero dk 16..31 ----
  {
    const int qr  = tid >> 1;
    const int dk0 = (tid & 1) << 3;
    const float* qp = q + (long long)(b * NTOK + qblk + qr) * DM + h * DKH + dk0;
    const float4 f0 = *(const float4*)(qp);
    const float4 f1 = *(const float4*)(qp + 4);
    v8h hv;
    hv[0] = (_Float16)f0.x; hv[1] = (_Float16)f0.y;
    hv[2] = (_Float16)f0.z; hv[3] = (_Float16)f0.w;
    hv[4] = (_Float16)f1.x; hv[5] = (_Float16)f1.y;
    hv[6] = (_Float16)f1.z; hv[7] = (_Float16)f1.w;
    *(v8h*)&Qt[qr][dk0]      = hv;
    *(v8h*)&Qt[qr][16 + dk0] = z8;
  }
  // zero the padded key range 16..31 of this wave's P tile (once)
  *(v8h*)&Pl[wave][lane & 15][16 + ((lane >> 4) << 3)] = z8;
  __syncthreads();

  const int qcol  = lane & 15;               // query column owned by this lane
  const int rbase = (lane < 16) ? 0 : 8;     // C-layout row base (keys / dk)
  const int k0h   = (lane < 16) ? 0 : 8;     // A-frag half select
  const int kb    = (lane < 16) ? 0 : 16;    // unused; B frags also use k0h
  const int qloc  = wave * 16 + qcol;        // query within block

  // per-lane query coords (lane and lane^16 duplicate the same query)
  const float* qcp = coords6 + (long long)(b * NTOK + qblk + qloc) * 6;
  const float qx = qcp[0], qy = qcp[1], qz = qcp[2];

  // B fragment for S^T (Q^T columns) -- loop-invariant
  const v16h qtf = cat16(*(const v8h*)&Qt[qloc][k0h],
                         *(const v8h*)&Qt[qloc][16 + k0h]);

  v8f acc;                                   // O^T: dk rows rbase..rbase+7
  #pragma unroll
  for (int r = 0; r < 8; ++r) acc[r] = 0.0f;
  float mrow = -3.0e38f, lrow = 0.0f;        // per-query scalars

  for (int kt = 0; kt < NTOK; kt += 16) {
    __syncthreads();
    {
      const int key = tid >> 4, dk = tid & 15;
      const long long gk = (long long)(b * NTOK + kt + key) * DM + h * DKH + dk;
      Kl[key][dk]       = (_Float16)k[gk];
      Kl[key][16 + dk]  = (_Float16)0.0f;
      Vtr[dk][key]      = (_Float16)v[gk];
      Vtr[dk][16 + key] = (_Float16)0.0f;
    }
    if (tid < 16) {
      const long long gk = (long long)(b * NTOK + kt + tid) * 6;
      kc4[tid] = make_float4(coords6[gk], coords6[gk + 1], coords6[gk + 2], 0.0f);
    }
    __syncthreads();

    // ---- S^T = K Q^T : A frag = K-tile rows (key = lane&15) ----
    const int arow = lane & 15;
    const v16h kaf = cat16(*(const v8h*)&Kl[arow][k0h],
                           *(const v8h*)&Kl[arow][16 + k0h]);
    v8f s8;
    #pragma unroll
    for (int r = 0; r < 8; ++r) s8[r] = 0.0f;
    s8 = wmma_f16(kaf, qtf, s8);

    // ---- geometric bias + per-lane online softmax over 8 keys ----
    float sc[8];
    float lm = -3.0e38f;
    #pragma unroll
    for (int r = 0; r < 8; ++r) {
      const float4 kcc = kc4[rbase + r];
      const float g = qx * kcc.x + qy * kcc.y + qz * kcc.z;
      const float s = s8[r] * 0.25f + alpha * g;   // 1/sqrt(16) = 0.25
      sc[r] = s;
      lm = fmaxf(lm, s);
    }
    const float rm = fmaxf(lm, __shfl_xor(lm, 16, 32));  // combine lane pair
    const float mn = fmaxf(mrow, rm);
    const float sf = __expf(mrow - mn);
    float ls = 0.0f;
    #pragma unroll
    for (int r = 0; r < 8; ++r) {
      const float p = __expf(sc[r] - mn);
      ls += p;
      Pl[wave][qcol][rbase + r] = (_Float16)p;     // P^T reg->LDS (A/B re-layout)
    }
    lrow = lrow * sf + ls + __shfl_xor(ls, 16, 32);
    mrow = mn;
    #pragma unroll
    for (int r = 0; r < 8; ++r) acc[r] *= sf;

    // ---- O^T += V^T P^T : A frag = Vtr rows (dk = lane&15) ----
    const v16h vaf = cat16(*(const v8h*)&Vtr[arow][k0h],
                           *(const v8h*)&Vtr[arow][16 + k0h]);
    const v16h pbf = cat16(*(const v8h*)&Pl[wave][qcol][k0h],
                           *(const v8h*)&Pl[wave][qcol][16 + k0h]);
    acc = wmma_f16(vaf, pbf, acc);
  }

  // ---- normalize + write: contiguous dk run of 8 -> two b128 stores ----
  const float inv = 1.0f / lrow;
  float4 o0, o1;
  o0.x = acc[0] * inv; o0.y = acc[1] * inv;
  o0.z = acc[2] * inv; o0.w = acc[3] * inv;
  o1.x = acc[4] * inv; o1.y = acc[5] * inv;
  o1.z = acc[6] * inv; o1.w = acc[7] * inv;
  float* op = aout + (long long)(b * NTOK + qblk + qloc) * DM + h * DKH + rbase;
  *(float4*)(op)     = o0;
  *(float4*)(op + 4) = o1;
}

// ---------------------------------------------------------------------------
// KNN: per point, top-16 nearest (self excluded), per batch.
// ---------------------------------------------------------------------------
__global__ void knn_kernel(const float* __restrict__ p6, int* __restrict__ idx)
{
  const int b = blockIdx.y;
  const int i = blockIdx.x * 256 + threadIdx.x;
  const float* base = p6 + (long long)b * NTOK * 6;
  const float xi = base[i * 6 + 0];
  const float yi = base[i * 6 + 1];
  const float zi = base[i * 6 + 2];

  float bd[KNN_K];
  int   bi[KNN_K];
  #pragma unroll
  for (int t = 0; t < KNN_K; ++t) { bd[t] = 3.0e38f; bi[t] = 0; }

  __shared__ float sx[256], sy[256], sz[256];
  for (int j0 = 0; j0 < NTOK; j0 += 256) {
    __syncthreads();
    const int j = j0 + threadIdx.x;
    sx[threadIdx.x] = base[j * 6 + 0];
    sy[threadIdx.x] = base[j * 6 + 1];
    sz[threadIdx.x] = base[j * 6 + 2];
    __syncthreads();
    for (int jj = 0; jj < 256; ++jj) {
      const int j2 = j0 + jj;
      if (j2 == i) continue;
      const float dx = xi - sx[jj], dy = yi - sy[jj], dz = zi - sz[jj];
      const float d2 = dx * dx + dy * dy + dz * dz;
      if (d2 < bd[KNN_K - 1]) {
        int p = KNN_K - 1;
        while (p > 0 && bd[p - 1] > d2) {
          bd[p] = bd[p - 1]; bi[p] = bi[p - 1]; --p;
        }
        bd[p] = d2; bi[p] = j2;
      }
    }
  }
  for (int t = 0; t < KNN_K; ++t)
    idx[((long long)b * NTOK + i) * KNN_K + t] = bi[t];
}

// GCN aggregate: out = relu((sum_k h[idx]+h)/17 + bias)
__global__ void gcn_agg_kernel(const float* __restrict__ h,
                               const int* __restrict__ idx,
                               const float* __restrict__ bias,
                               float* __restrict__ out, int C)
{
  const long long t = (long long)blockIdx.x * 256 + threadIdx.x;
  const long long total = (long long)NB * NTOK * C;
  if (t >= total) return;
  const int c = (int)(t % C);
  const long long bn = t / C;
  const int n = (int)(bn % NTOK);
  const int b = (int)(bn / NTOK);
  float accum = h[t];
  const int* ip = idx + ((long long)b * NTOK + n) * KNN_K;
  #pragma unroll
  for (int kk = 0; kk < KNN_K; ++kk)
    accum += h[((long long)b * NTOK + ip[kk]) * C + c];
  float vv = accum * (1.0f / (KNN_K + 1)) + bias[c];
  out[t] = fmaxf(vv, 0.0f);
}

// x = LN(x + add) * g + b ; one 128-thread block per row.
__global__ void ln_res_kernel(float* __restrict__ x,
                              const float* __restrict__ add,
                              const float* __restrict__ g,
                              const float* __restrict__ bb)
{
  const int row = blockIdx.x;
  const int c = threadIdx.x;
  __shared__ float red[128];
  const long long o = (long long)row * DM + c;
  const float val = x[o] + add[o];
  red[c] = val; __syncthreads();
  for (int s = 64; s > 0; s >>= 1) { if (c < s) red[c] += red[c + s]; __syncthreads(); }
  const float mean = red[0] * (1.0f / DM);
  __syncthreads();
  const float d = val - mean;
  red[c] = d * d; __syncthreads();
  for (int s = 64; s > 0; s >>= 1) { if (c < s) red[c] += red[c + s]; __syncthreads(); }
  const float var = red[0] * (1.0f / DM);
  x[o] = d * rsqrtf(var + 1e-5f) * g[c] + bb[c];
}

__global__ void mean_kernel(const float* __restrict__ x, float* __restrict__ gf)
{
  const int b = blockIdx.x, c = threadIdx.x;
  float accum = 0.0f;
  for (int n = 0; n < NTOK; ++n)
    accum += x[((long long)b * NTOK + n) * DM + c];
  gf[b * DM + c] = accum * (1.0f / NTOK);
}

// pcd = (relu(gfeat@w1+b1)@w2+b2) -> (B,3,64) ; one block per batch.
__global__ void init_fc_kernel(const float* __restrict__ gf,
                               const float* __restrict__ w1, const float* __restrict__ b1,
                               const float* __restrict__ w2, const float* __restrict__ b2,
                               float* __restrict__ pcd)
{
  const int b = blockIdx.x, t = threadIdx.x;
  __shared__ float sg[DM], sh[DM];
  if (t < DM) sg[t] = gf[b * DM + t];
  __syncthreads();
  if (t < DM) {
    float accum = b1[t];
    for (int kk = 0; kk < DM; ++kk) accum += sg[kk] * w1[kk * DM + t];
    sh[t] = fmaxf(accum, 0.0f);
  }
  __syncthreads();
  if (t < 3 * NCOARSE) {
    float accum = b2[t];
    for (int kk = 0; kk < DM; ++kk) accum += sh[kk] * w2[kk * (3 * NCOARSE) + t];
    pcd[(long long)b * 3 * NCOARSE + t] = accum;
  }
}

__global__ void fill0_kernel(float* __restrict__ p, long long n)
{
  const long long t = (long long)blockIdx.x * 256 + threadIdx.x;
  if (t < n) p[t] = 0.0f;
}

// local[b][o][i] = pcd_w[o,:3] . pcd[b,:,i] + pcd_b[o]
__global__ void pcd_local_kernel(const float* __restrict__ w,
                                 const float* __restrict__ bia,
                                 const float* __restrict__ pcd,
                                 float* __restrict__ local, int n)
{
  const long long t = (long long)blockIdx.x * 256 + threadIdx.x;
  const long long total = (long long)NB * 64 * n;
  if (t >= total) return;
  const int i = (int)(t % n);
  const int o = (int)((t / n) % 64);
  const int b = (int)(t / ((long long)64 * n));
  const float* pc = pcd + (long long)b * 3 * n;
  float accum = bia[o];
  accum += w[o * 3 + 0] * pc[0 * n + i];
  accum += w[o * 3 + 1] * pc[1 * n + i];
  accum += w[o * 3 + 2] * pc[2 * n + i];
  local[t] = accum;
}

// cat[b][0:64] = local ; cat[b][64:192] = Kf
__global__ void concat_kernel(const float* __restrict__ local,
                              const float* __restrict__ Kf,
                              float* __restrict__ cat, int n)
{
  const long long t = (long long)blockIdx.x * 256 + threadIdx.x;
  const long long total = (long long)NB * 192 * n;
  if (t >= total) return;
  const int i = (int)(t % n);
  const int ch = (int)((t / n) % 192);
  const int b = (int)(t / ((long long)192 * n));
  float vv;
  if (ch < 64) vv = local[((long long)b * 64 + ch) * n + i];
  else         vv = Kf[((long long)b * DM + (ch - 64)) * n + i];
  cat[t] = vv;
}

// pcd_out[b][c][j] = pcd_in[b][c][j/2] + tanh(co_w[c,:]@off[b,:,j] + co_b[c])
__global__ void co_update_kernel(const float* __restrict__ co_w,
                                 const float* __restrict__ co_b,
                                 const float* __restrict__ off,
                                 const float* __restrict__ pcd_in,
                                 float* __restrict__ pcd_out,
                                 int n_in, int n_out)
{
  const long long t = (long long)blockIdx.x * 256 + threadIdx.x;
  const long long total = (long long)NB * n_out;
  if (t >= total) return;
  const int j = (int)(t % n_out);
  const int b = (int)(t / n_out);
  const int i = j >> 1;                       // uf == 2 for all stages
  const float* ob = off + (long long)b * DM * n_out;
  #pragma unroll
  for (int c = 0; c < 3; ++c) {
    float accum = co_b[c];
    for (int kk = 0; kk < DM; ++kk)
      accum += co_w[c * DM + kk] * ob[(long long)kk * n_out + j];
    pcd_out[((long long)b * 3 + c) * n_out + j] =
        pcd_in[((long long)b * 3 + c) * n_in + i] + tanhf(accum);
  }
}

// ---------------------------------------------------------------------------
// Host side
// ---------------------------------------------------------------------------
struct LayerP {
  const float *wq,*bq,*wk,*bk,*wv,*bv,*alpha,*ffw1,*ffb1,*ffw2,*ffb2,
              *ln1g,*ln1b,*ln2g,*ln2b;
};
struct StageP {
  const float *pcd_w,*pcd_b,*merge_w,*merge_b,*dec_w,*dec_b,
              *ow1,*ob1,*ow2,*ob2,*ows,*obs,*co_w,*co_b;
};

static void gemm_launch(hipStream_t stream,
    const float* A, int asr, int asc, long long ab,
    const float* Bp, int bsr, int bsc, long long bb,
    float* C, int csr, int csc, long long cb, int coff,
    const float* biasCol, const float* biasRow,
    const float* rowAdd, long long rab,
    int M, int Nn, int K, int act, int accum, int batch)
{
  dim3 g((Nn + 63) / 64, (M + 63) / 64, batch);
  const bool guarded = (M % 64) || (Nn % 64) || (K % 32);
  if (guarded)
    wmma_gemm_kernel<true><<<g, 256, 0, stream>>>(
        A, asr, asc, ab, Bp, bsr, bsc, bb, C, csr, csc, cb, coff,
        biasCol, biasRow, rowAdd, rab, M, Nn, K, act, accum);
  else
    wmma_gemm_kernel<false><<<g, 256, 0, stream>>>(
        A, asr, asc, ab, Bp, bsr, bsc, bb, C, csr, csc, cb, coff,
        biasCol, biasRow, rowAdd, rab, M, Nn, K, act, accum);
}

extern "C" void kernel_launch(void* const* d_in, const int* in_sizes, int n_in,
                              void* d_out, int out_size, void* d_ws, size_t ws_size,
                              hipStream_t stream)
{
  (void)out_size; (void)ws_size;

  // ---- parameter mapping (detect dict-insertion vs jax-sorted flattening) ---
  int li = 0;
  auto nxt = [&]() { return (const float*)d_in[li++]; };

  const float *partial, *gw0, *gb0, *gw1, *gb1, *finw, *finb;
  const float *ifc_w1, *ifc_b1, *ifc_w2, *ifc_b2;
  LayerP L[NLAY];
  StageP S[NSTAGE];

  const bool insertion = (n_in > 0) && (in_sizes[0] == NB * NTOK * 6);
  if (insertion) {
    partial = nxt();
    gw0 = nxt(); gb0 = nxt(); gw1 = nxt(); gb1 = nxt();
    finw = nxt(); finb = nxt();
    for (int l = 0; l < NLAY; ++l) {
      L[l].wq = nxt(); L[l].bq = nxt(); L[l].wk = nxt(); L[l].bk = nxt();
      L[l].wv = nxt(); L[l].bv = nxt(); L[l].alpha = nxt();
      L[l].ffw1 = nxt(); L[l].ffb1 = nxt(); L[l].ffw2 = nxt(); L[l].ffb2 = nxt();
      L[l].ln1g = nxt(); L[l].ln1b = nxt(); L[l].ln2g = nxt(); L[l].ln2b = nxt();
    }
    ifc_w1 = nxt(); ifc_b1 = nxt(); ifc_w2 = nxt(); ifc_b2 = nxt();
    for (int s = 0; s < NSTAGE; ++s) {
      S[s].pcd_w = nxt(); S[s].pcd_b = nxt();
      S[s].merge_w = nxt(); S[s].merge_b = nxt();
      S[s].dec_w = nxt(); S[s].dec_b = nxt();
      S[s].ow1 = nxt(); S[s].ob1 = nxt(); S[s].ow2 = nxt(); S[s].ob2 = nxt();
      S[s].ows = nxt(); S[s].obs = nxt(); S[s].co_w = nxt(); S[s].co_b = nxt();
    }
  } else {
    // jax tree order: dict keys sorted at every level; 'params' < 'partial_6d'
    finb = nxt(); finw = nxt();                         // final_lin: b, w
    gb0 = nxt(); gw0 = nxt(); gb1 = nxt(); gw1 = nxt(); // gcn list of {b,w}
    ifc_b1 = nxt(); ifc_b2 = nxt(); ifc_w1 = nxt(); ifc_w2 = nxt();
    for (int l = 0; l < NLAY; ++l) {
      L[l].alpha = nxt(); L[l].bk = nxt(); L[l].bq = nxt(); L[l].bv = nxt();
      L[l].ffb1 = nxt(); L[l].ffb2 = nxt(); L[l].ffw1 = nxt(); L[l].ffw2 = nxt();
      L[l].ln1b = nxt(); L[l].ln1g = nxt(); L[l].ln2b = nxt(); L[l].ln2g = nxt();
      L[l].wk = nxt(); L[l].wq = nxt(); L[l].wv = nxt();
    }
    for (int s = 0; s < NSTAGE; ++s) {
      S[s].co_b = nxt(); S[s].co_w = nxt();
      S[s].dec_b = nxt(); S[s].dec_w = nxt();
      S[s].merge_b = nxt(); S[s].merge_w = nxt();
      S[s].ob1 = nxt(); S[s].ob2 = nxt(); S[s].obs = nxt();
      S[s].ow1 = nxt(); S[s].ow2 = nxt(); S[s].ows = nxt();
      S[s].pcd_b = nxt(); S[s].pcd_w = nxt();
    }
    partial = nxt();
  }

  // ---- workspace bump allocator (256B aligned) ----
  char* wp = (char*)d_ws;
  auto alloc = [&](size_t bytes) {
    void* p = (void*)wp;
    wp += (bytes + 255) & ~(size_t)255;
    return p;
  };
  const size_t TOKF = (size_t)NB * NTOK;            // 4096 token rows
  int*   idxb  = (int*)  alloc(TOKF * KNN_K * 4);
  float* h0    = (float*)alloc(TOKF * 64 * 4);
  float* x1    = (float*)alloc(TOKF * 64 * 4);
  float* hbuf  = (float*)alloc(TOKF * DM * 4);
  float* xtmp  = (float*)alloc(TOKF * DM * 4);
  float* xb    = (float*)alloc(TOKF * DM * 4);
  float* qb    = (float*)alloc(TOKF * DM * 4);
  float* kb    = (float*)alloc(TOKF * DM * 4);
  float* vb    = (float*)alloc(TOKF * DM * 4);
  float* abuf  = (float*)alloc(TOKF * DM * 4);
  float* ffh   = (float*)alloc(TOKF * 512 * 4);
  float* ffo   = (float*)alloc(TOKF * DM * 4);
  float* gfeat = (float*)alloc((size_t)NB * DM * 4);
  float* pcdA  = (float*)alloc((size_t)NB * 3 * 8192 * 4);
  float* pcdB  = (float*)alloc((size_t)NB * 3 * 8192 * 4);
  float* localb= (float*)alloc((size_t)NB * 64 * 4096 * 4);
  float* catb  = (float*)alloc((size_t)NB * 192 * 4096 * 4);
  float* feats = (float*)alloc((size_t)NB * DM * 4096 * 4);
  float* kup   = (float*)alloc((size_t)NB * DM * 8192 * 4);
  float* h1s   = (float*)alloc((size_t)NB * DM * 8192 * 4);
  float* offA  = (float*)alloc((size_t)NB * DM * 8192 * 4);
  float* offB  = (float*)alloc((size_t)NB * DM * 8192 * 4);

  // ================= Feature extraction =================
  knn_kernel<<<dim3(NTOK / 256, NB), 256, 0, stream>>>(partial, idxb);

  // h0 = partial(4096x6) @ w0(6x64)  (ragged K=6 -> guarded variant)
  gemm_launch(stream, partial, 6, 1, 0, gw0, 64, 1, 0,
              h0, 64, 1, 0, 0, nullptr, nullptr, nullptr, 0,
              (int)TOKF, 64, 6, 0, 0, 1);
  {
    long long tot = (long long)TOKF * 64;
    gcn_agg_kernel<<<(unsigned)((tot + 255) / 256), 256, 0, stream>>>(h0, idxb, gb0, x1, 64);
  }
  gemm_launch(stream, x1, 64, 1, 0, gw1, DM, 1, 0,
              hbuf, DM, 1, 0, 0, nullptr, nullptr, nullptr, 0,
              (int)TOKF, DM, 64, 0, 0, 1);
  {
    long long tot = (long long)TOKF * DM;
    gcn_agg_kernel<<<(unsigned)((tot + 255) / 256), 256, 0, stream>>>(hbuf, idxb, gb1, xtmp, DM);
  }
  gemm_launch(stream, xtmp, DM, 1, 0, finw, DM, 1, 0,
              xb, DM, 1, 0, 0, finb, nullptr, nullptr, 0,
              (int)TOKF, DM, DM, 0, 0, 1);

  // ================= Transformer =================
  for (int l = 0; l < NLAY; ++l) {
    gemm_launch(stream, xb, DM, 1, 0, L[l].wq, DM, 1, 0, qb, DM, 1, 0, 0,
                L[l].bq, nullptr, nullptr, 0, (int)TOKF, DM, DM, 0, 0, 1);
    gemm_launch(stream, xb, DM, 1, 0, L[l].wk, DM, 1, 0, kb, DM, 1, 0, 0,
                L[l].bk, nullptr, nullptr, 0, (int)TOKF, DM, DM, 0, 0, 1);
    gemm_launch(stream, xb, DM, 1, 0, L[l].wv, DM, 1, 0, vb, DM, 1, 0, 0,
                L[l].bv, nullptr, nullptr, 0, (int)TOKF, DM, DM, 0, 0, 1);
    attn_kernel<<<dim3(NTOK / 128, NH, NB), 256, 0, stream>>>(
        qb, kb, vb, partial, L[l].alpha, abuf);
    ln_res_kernel<<<(unsigned)TOKF, DM, 0, stream>>>(xb, abuf, L[l].ln1g, L[l].ln1b);
    gemm_launch(stream, xb, DM, 1, 0, L[l].ffw1, 512, 1, 0, ffh, 512, 1, 0, 0,
                L[l].ffb1, nullptr, nullptr, 0, (int)TOKF, 512, DM, 1, 0, 1);
    gemm_launch(stream, ffh, 512, 1, 0, L[l].ffw2, DM, 1, 0, ffo, DM, 1, 0, 0,
                L[l].ffb2, nullptr, nullptr, 0, (int)TOKF, DM, 512, 0, 0, 1);
    ln_res_kernel<<<(unsigned)TOKF, DM, 0, stream>>>(xb, ffo, L[l].ln2g, L[l].ln2b);
  }

  // ================= Decoder init =================
  mean_kernel<<<NB, DM, 0, stream>>>(xb, gfeat);
  init_fc_kernel<<<NB, 192, 0, stream>>>(gfeat, ifc_w1, ifc_b1, ifc_w2, ifc_b2, pcdA);
  {
    long long tot = (long long)NB * DM * NCOARSE;
    fill0_kernel<<<(unsigned)((tot + 255) / 256), 256, 0, stream>>>(offA, tot);
  }

  // ================= Snowflake stages =================
  float* pcd_in = pcdA; float* pcd_out = pcdB;
  float* Kf = offA;     float* off = offB;
  int n = NCOARSE;
  for (int s = 0; s < NSTAGE; ++s) {
    const StageP& sp = S[s];
    const int no = n * 2;
    {
      long long tot = (long long)NB * 64 * n;
      pcd_local_kernel<<<(unsigned)((tot + 255) / 256), 256, 0, stream>>>(
          sp.pcd_w, sp.pcd_b, pcd_in, localb, n);
      tot = (long long)NB * 192 * n;
      concat_kernel<<<(unsigned)((tot + 255) / 256), 256, 0, stream>>>(
          localb, Kf, catb, n);
    }
    // feats = merge_w(128x192) @ cat + merge_b + gcol
    gemm_launch(stream, sp.merge_w, 192, 1, 0,
                catb, n, 1, (long long)192 * n,
                feats, n, 1, (long long)DM * n, 0,
                nullptr, sp.merge_b, gfeat, DM,
                DM, n, 192, 0, 0, NB);
    // K_up[b,o,i*2+t] = sum_c feats[b,c,i] * dec_w[c,o,t]  (+ dec_b[o])
    for (int t = 0; t < 2; ++t)
      gemm_launch(stream, sp.dec_w + t, 2, DM * 2, 0,
                  feats, n, 1, (long long)DM * n,
                  kup, no, 2, (long long)DM * no, t,
                  nullptr, sp.dec_b, nullptr, 0,
                  DM, n, DM, 0, 0, NB);
    // h1 = relu(off_w1 @ K_up + off_b1)
    gemm_launch(stream, sp.ow1, DM, 1, 0,
                kup, no, 1, (long long)DM * no,
                h1s, no, 1, (long long)DM * no, 0,
                nullptr, sp.ob1, nullptr, 0,
                DM, no, DM, 1, 0, NB);
    // off = off_w2 @ h1 + off_b2
    gemm_launch(stream, sp.ow2, DM, 1, 0,
                h1s, no, 1, (long long)DM * no,
                off, no, 1, (long long)DM * no, 0,
                nullptr, sp.ob2, nullptr, 0,
                DM, no, DM, 0, 0, NB);
    // off += off_ws @ K_up + off_bs
    gemm_launch(stream, sp.ows, DM, 1, 0,
                kup, no, 1, (long long)DM * no,
                off, no, 1, (long long)DM * no, 0,
                nullptr, sp.obs, nullptr, 0,
                DM, no, DM, 0, 1, NB);
    // pcd update (radius==1.0 so /radius**s is identity)
    float* pout = (s == NSTAGE - 1) ? (float*)d_out : pcd_out;
    {
      long long tot = (long long)NB * no;
      co_update_kernel<<<(unsigned)((tot + 255) / 256), 256, 0, stream>>>(
          sp.co_w, sp.co_b, off, pcd_in, pout, n, no);
    }
    float* tp = pcd_in; pcd_in = pcd_out; pcd_out = tp;
    tp = Kf; Kf = off; off = tp;
    n = no;
  }
}
